// AdaptiveProteinBlock_86380382257292
// MI455X (gfx1250) — matrix operand order repaired and checked by reading, not compiled, not run on hardware
//
#include <hip/hip_runtime.h>
#include <hip/hip_bf16.h>
#include <cstdint>
#include <cstddef>

#define NROWS 4096
#define DIN   512
#define DATT  64
#define NSTEPS 2
#define PNUC  0.9f
#define LNEPS 1e-5f

typedef __bf16 bf16;
typedef __attribute__((ext_vector_type(16))) __bf16 v16bf;
typedef __attribute__((ext_vector_type(8)))  __bf16 v8bf;
typedef __attribute__((ext_vector_type(8)))  float  v8f;

union FragU { v16bf v; v8bf h[2]; };

// ---------------------------------------------------------------------------
// Fragment layouts (wave32, V_WMMA_F32_16X16X32_BF16):
// A (16x32): lane = 16h + m ; elems 0..7 -> K = kb+8h+e ; 8..15 -> K = kb+16+8h+(e-8)
// B (32x16): lane = 16h + n ; elems 0..7 -> K = kb+8h+e ; 8..15 -> K = kb+16+8h+(e-8)
// ---------------------------------------------------------------------------
__device__ __forceinline__ v16bf load_fragA(const bf16* __restrict__ base, int ld,
                                            int row0, int kb, int lane) {
  int r = row0 + (lane & 15);
  int h = lane >> 4;
  const bf16* p = base + (size_t)r * ld + kb + 8 * h;
  FragU f;
  f.h[0] = *(const v8bf*)(p);
  f.h[1] = *(const v8bf*)(p + 16);
  return f.v;
}

__device__ __forceinline__ v16bf load_fragB(const bf16* __restrict__ base, int ld,
                                            int col0, int kb, int lane) {
  int n = col0 + (lane & 15);
  int h = lane >> 4;
  const bf16* p = base + (size_t)n * ld + kb + 8 * h;
  FragU f;
  f.h[0] = *(const v8bf*)(p);
  f.h[1] = *(const v8bf*)(p + 16);
  return f.v;
}

// ---------------------------------------------------------------------------
// Direct-global WMMA GEMM for small-N shapes (attention projections, N=64).
// Block = 128 threads = 4 waves; wave tile 32x64; grid = (N/64, M/128).
// ---------------------------------------------------------------------------
__global__ void __launch_bounds__(128)
gemm_bf16(const bf16* __restrict__ A, int lda,
          const bf16* __restrict__ Bt, int ldb,
          float* __restrict__ C, int ldc,
          int K, int accumulate) {
  const int lane = threadIdx.x & 31;
  const int wave = threadIdx.x >> 5;
  const int row0 = blockIdx.y * 128 + wave * 32;
  const int col0 = blockIdx.x * 64;
  const int n = lane & 15;
  const int h = lane >> 4;

  v8f acc[2][4];
  if (accumulate) {
#pragma unroll
    for (int mi = 0; mi < 2; ++mi)
#pragma unroll
      for (int nj = 0; nj < 4; ++nj) {
        int mr0 = row0 + 16 * mi + 8 * h;
        int nc  = col0 + 16 * nj + n;
#pragma unroll
        for (int r = 0; r < 8; ++r)
          acc[mi][nj][r] = C[(size_t)(mr0 + r) * ldc + nc];
      }
  } else {
#pragma unroll
    for (int mi = 0; mi < 2; ++mi)
#pragma unroll
      for (int nj = 0; nj < 4; ++nj)
#pragma unroll
        for (int r = 0; r < 8; ++r)
          acc[mi][nj][r] = 0.0f;
  }

  for (int kb = 0; kb < K; kb += 32) {
    v16bf a0 = load_fragA(A, lda, row0,      kb, lane);
    v16bf a1 = load_fragA(A, lda, row0 + 16, kb, lane);
    v16bf b0 = load_fragB(Bt, ldb, col0,      kb, lane);
    v16bf b1 = load_fragB(Bt, ldb, col0 + 16, kb, lane);
    v16bf b2 = load_fragB(Bt, ldb, col0 + 32, kb, lane);
    v16bf b3 = load_fragB(Bt, ldb, col0 + 48, kb, lane);
    acc[0][0] = __builtin_amdgcn_wmma_f32_16x16x32_bf16(false, a0, false, b0, (short)0, acc[0][0], false, false);
    acc[0][1] = __builtin_amdgcn_wmma_f32_16x16x32_bf16(false, a0, false, b1, (short)0, acc[0][1], false, false);
    acc[0][2] = __builtin_amdgcn_wmma_f32_16x16x32_bf16(false, a0, false, b2, (short)0, acc[0][2], false, false);
    acc[0][3] = __builtin_amdgcn_wmma_f32_16x16x32_bf16(false, a0, false, b3, (short)0, acc[0][3], false, false);
    acc[1][0] = __builtin_amdgcn_wmma_f32_16x16x32_bf16(false, a1, false, b0, (short)0, acc[1][0], false, false);
    acc[1][1] = __builtin_amdgcn_wmma_f32_16x16x32_bf16(false, a1, false, b1, (short)0, acc[1][1], false, false);
    acc[1][2] = __builtin_amdgcn_wmma_f32_16x16x32_bf16(false, a1, false, b2, (short)0, acc[1][2], false, false);
    acc[1][3] = __builtin_amdgcn_wmma_f32_16x16x32_bf16(false, a1, false, b3, (short)0, acc[1][3], false, false);
  }

#pragma unroll
  for (int mi = 0; mi < 2; ++mi)
#pragma unroll
    for (int nj = 0; nj < 4; ++nj) {
      int mr0 = row0 + 16 * mi + 8 * h;
      int nc  = col0 + 16 * nj + n;
#pragma unroll
      for (int r = 0; r < 8; ++r)
        C[(size_t)(mr0 + r) * ldc + nc] = acc[mi][nj][r];
    }
}

// ---------------------------------------------------------------------------
// LDS-staged WMMA GEMM with CDNA5 async memory->LDS copies (ASYNCcnt tracked).
// Block = 256 threads = 8 waves (4x2); block tile 128x128; wave tile 32x64.
// Double-buffered 2 x (8KB A + 8KB B) LDS. grid = (N/128, M/128); K % 32 == 0.
// ---------------------------------------------------------------------------
#define GBM 128
#define GBN 128
#define GBK 32

__device__ __forceinline__ void async_b128_to_lds(uint32_t lds_addr, const void* gaddr) {
  asm volatile("global_load_async_to_lds_b128 %0, %1, off"
               :: "v"(lds_addr), "v"(gaddr) : "memory");
}

__global__ void __launch_bounds__(256)
gemm_bf16_lds(const bf16* __restrict__ A, int lda,
              const bf16* __restrict__ Bt, int ldb,
              float* __restrict__ C, int ldc,
              int K, int accumulate) {
  __shared__ bf16 As[2][GBM * GBK];   // 8 KB per buffer
  __shared__ bf16 Bs[2][GBN * GBK];   // 8 KB per buffer

  const int tid  = threadIdx.x;
  const int lane = tid & 31;
  const int wave = tid >> 5;
  const int wm = wave >> 1;                 // 0..3
  const int wn = wave & 1;                  // 0..1
  const int rowB = blockIdx.y * GBM;
  const int colB = blockIdx.x * GBN;
  const int row0 = wm * 32;                 // wave row offset within block tile
  const int col0 = wn * 64;                 // wave col offset within block tile
  const int n = lane & 15;
  const int h = lane >> 4;

  v8f acc[2][4];
  if (accumulate) {
#pragma unroll
    for (int mi = 0; mi < 2; ++mi)
#pragma unroll
      for (int nj = 0; nj < 4; ++nj) {
        int mr0 = rowB + row0 + 16 * mi + 8 * h;
        int nc  = colB + col0 + 16 * nj + n;
#pragma unroll
        for (int r = 0; r < 8; ++r)
          acc[mi][nj][r] = C[(size_t)(mr0 + r) * ldc + nc];
      }
  } else {
#pragma unroll
    for (int mi = 0; mi < 2; ++mi)
#pragma unroll
      for (int nj = 0; nj < 4; ++nj)
#pragma unroll
        for (int r = 0; r < 8; ++r)
          acc[mi][nj][r] = 0.0f;
  }

  // Cooperative async stage of one 128x32 A slab + 128x32 Bt slab.
  // 512 chunks of 16B per slab -> each of 256 threads issues 2 A + 2 B asyncs.
  auto stage = [&](int buf, int kb) {
#pragma unroll
    for (int j = 0; j < 2; ++j) {
      int t = tid + 256 * j;
      int r = t >> 2;
      int c = (t & 3) * 8;
      const bf16* ga = A  + (size_t)(rowB + r) * lda + kb + c;
      const bf16* gb = Bt + (size_t)(colB + r) * ldb + kb + c;
      uint32_t la = (uint32_t)(uintptr_t)(&As[buf][r * GBK + c]);
      uint32_t lb = (uint32_t)(uintptr_t)(&Bs[buf][r * GBK + c]);
      async_b128_to_lds(la, (const void*)ga);
      async_b128_to_lds(lb, (const void*)gb);
    }
  };

  stage(0, 0);
  int buf = 0;
  for (int kb = 0; kb < K; kb += GBK) {
    asm volatile("s_wait_asynccnt 0" ::: "memory");
    __syncthreads();
    if (kb + GBK < K) stage(buf ^ 1, kb + GBK);

    // A fragments from LDS (row-major [128][32] slab, kb-local K origin 0)
    const bf16* a0p = &As[buf][(row0      + (lane & 15)) * GBK + 8 * h];
    const bf16* a1p = &As[buf][(row0 + 16 + (lane & 15)) * GBK + 8 * h];
    FragU fa0, fa1;
    fa0.h[0] = *(const v8bf*)(a0p);  fa0.h[1] = *(const v8bf*)(a0p + 16);
    fa1.h[0] = *(const v8bf*)(a1p);  fa1.h[1] = *(const v8bf*)(a1p + 16);

    v16bf b[4];
#pragma unroll
    for (int nj = 0; nj < 4; ++nj) {
      const bf16* bp = &Bs[buf][(col0 + 16 * nj + (lane & 15)) * GBK + 8 * h];
      FragU fb;
      fb.h[0] = *(const v8bf*)(bp);
      fb.h[1] = *(const v8bf*)(bp + 16);
      b[nj] = fb.v;
    }

    acc[0][0] = __builtin_amdgcn_wmma_f32_16x16x32_bf16(false, fa0.v, false, b[0], (short)0, acc[0][0], false, false);
    acc[0][1] = __builtin_amdgcn_wmma_f32_16x16x32_bf16(false, fa0.v, false, b[1], (short)0, acc[0][1], false, false);
    acc[0][2] = __builtin_amdgcn_wmma_f32_16x16x32_bf16(false, fa0.v, false, b[2], (short)0, acc[0][2], false, false);
    acc[0][3] = __builtin_amdgcn_wmma_f32_16x16x32_bf16(false, fa0.v, false, b[3], (short)0, acc[0][3], false, false);
    acc[1][0] = __builtin_amdgcn_wmma_f32_16x16x32_bf16(false, fa1.v, false, b[0], (short)0, acc[1][0], false, false);
    acc[1][1] = __builtin_amdgcn_wmma_f32_16x16x32_bf16(false, fa1.v, false, b[1], (short)0, acc[1][1], false, false);
    acc[1][2] = __builtin_amdgcn_wmma_f32_16x16x32_bf16(false, fa1.v, false, b[2], (short)0, acc[1][2], false, false);
    acc[1][3] = __builtin_amdgcn_wmma_f32_16x16x32_bf16(false, fa1.v, false, b[3], (short)0, acc[1][3], false, false);

    buf ^= 1;
  }

#pragma unroll
  for (int mi = 0; mi < 2; ++mi)
#pragma unroll
    for (int nj = 0; nj < 4; ++nj) {
      int mr0 = rowB + row0 + 16 * mi + 8 * h;
      int nc  = colB + col0 + 16 * nj + n;
#pragma unroll
      for (int r = 0; r < 8; ++r)
        C[(size_t)(mr0 + r) * ldc + nc] = acc[mi][nj][r];
    }
}

// ---------------------------------------------------------------------------
// Reductions (blockDim = 256 -> 8 waves of 32)
// ---------------------------------------------------------------------------
__device__ __forceinline__ float block_sum(float v, float* red) {
#pragma unroll
  for (int off = 16; off > 0; off >>= 1) v += __shfl_down(v, off, 32);
  int lane = threadIdx.x & 31, w = threadIdx.x >> 5;
  if (lane == 0) red[w] = v;
  __syncthreads();
  if (w == 0) {
    float t = (lane < 8) ? red[lane] : 0.f;
#pragma unroll
    for (int off = 4; off > 0; off >>= 1) t += __shfl_down(t, off, 32);
    if (lane == 0) red[0] = t;
  }
  __syncthreads();
  float r = red[0];
  __syncthreads();
  return r;
}

__device__ __forceinline__ float block_max(float v, float* red) {
#pragma unroll
  for (int off = 16; off > 0; off >>= 1) v = fmaxf(v, __shfl_down(v, off, 32));
  int lane = threadIdx.x & 31, w = threadIdx.x >> 5;
  if (lane == 0) red[w] = v;
  __syncthreads();
  if (w == 0) {
    float t = (lane < 8) ? red[lane] : -3.4e38f;
#pragma unroll
    for (int off = 4; off > 0; off >>= 1) t = fmaxf(t, __shfl_down(t, off, 32));
    if (lane == 0) red[0] = t;
  }
  __syncthreads();
  float r = red[0];
  __syncthreads();
  return r;
}

// ---------------------------------------------------------------------------
// Prep kernels
// ---------------------------------------------------------------------------
__global__ void rowprep(const float* __restrict__ S, bf16* __restrict__ Sb,
                        float* __restrict__ rowinv) {
  __shared__ float red[8];
  int i = blockIdx.x;
  float s = 0.f;
  for (int j = threadIdx.x; j < NROWS; j += blockDim.x) {
    float v = S[(size_t)i * NROWS + j] + (j == i ? 1.f : 0.f);
    Sb[(size_t)i * NROWS + j] = (bf16)v;
    s += v;
  }
  s = block_sum(s, red);
  if (threadIdx.x == 0) rowinv[i] = 1.f / fmaxf(s, 1e-12f);
}

__global__ void colprep(const float* __restrict__ S, float* __restrict__ colinv) {
  int j = blockIdx.x * blockDim.x + threadIdx.x;
  if (j < NROWS) {
    float s = 1.f;  // +I contributes 1 on the diagonal of column j
    for (int i = 0; i < NROWS; ++i) s += S[(size_t)i * NROWS + j];
    colinv[j] = 1.f / fmaxf(s, 1e-12f);
  }
}

__global__ void init_x(const float* __restrict__ X,
                       bf16* Xf, bf16* Xb, bf16* Xa,
                       bf16* Xft, bf16* Xbt, bf16* Xat,
                       float* __restrict__ Z) {
  int i = blockIdx.x;
  for (int d = threadIdx.x; d < DIN; d += blockDim.x) {
    float v = X[(size_t)i * DIN + d];
    bf16 b = (bf16)v;
    size_t rm = (size_t)i * DIN + d;
    size_t tm = (size_t)d * NROWS + i;
    Xf[rm] = b; Xb[rm] = b; Xa[rm] = b;
    Xft[tm] = b; Xbt[tm] = b; Xat[tm] = b;
    Z[rm] = 0.f;
  }
}

__global__ void tconv_f32_bf16(const float* __restrict__ src, bf16* __restrict__ dst,
                               int R, int C) {
  int i = blockIdx.x * blockDim.x + threadIdx.x;
  if (i < R * C) {
    int r = i / C, c = i % C;
    dst[(size_t)c * R + r] = (bf16)src[i];
  }
}

__global__ void conv_f32_bf16(const float* __restrict__ src, bf16* __restrict__ dst, int n) {
  int i = blockIdx.x * blockDim.x + threadIdx.x;
  if (i < n) dst[i] = (bf16)src[i];
}

__global__ void scale_conv(const float* __restrict__ T, const float* __restrict__ scale,
                           bf16* __restrict__ Xrm, bf16* __restrict__ Xt) {
  int i = blockIdx.x;
  float sc = scale ? scale[i] : 1.f;
  for (int d = threadIdx.x; d < DIN; d += blockDim.x) {
    float v = T[(size_t)i * DIN + d] * sc;
    bf16 b = (bf16)v;
    Xrm[(size_t)i * DIN + d] = b;
    Xt[(size_t)d * NROWS + i] = b;
  }
}

// ---------------------------------------------------------------------------
// Nucleus (top-p) sparse softmax over one row of logits (threshold by binary
// search on probability: same kept-set as rank-based nucleus for tie-free floats)
// ---------------------------------------------------------------------------
__global__ void nucleus_kernel(const float* __restrict__ Lg, bf16* __restrict__ Ab) {
  __shared__ float prow[NROWS];
  __shared__ float red[8];
  int i = blockIdx.x;
  const float* L = Lg + (size_t)i * NROWS;

  float mx = -3.4e38f;
  for (int j = threadIdx.x; j < NROWS; j += blockDim.x) {
    float v = L[j];
    prow[j] = v;
    mx = fmaxf(mx, v);
  }
  mx = block_max(mx, red);

  float tot = 0.f;
  for (int j = threadIdx.x; j < NROWS; j += blockDim.x) {
    float p = __expf(prow[j] - mx);
    prow[j] = p;
    tot += p;
  }
  tot = block_sum(tot, red);
  float target = PNUC * tot;

  float lo = 0.f, hi = 1.0f;   // max prob is exactly 1 (argmax element)
  float kept = tot;
  for (int it = 0; it < 30; ++it) {
    float mid = 0.5f * (lo + hi);
    float m = 0.f;
    for (int j = threadIdx.x; j < NROWS; j += blockDim.x) {
      float p = prow[j];
      m += (p >= mid) ? p : 0.f;
    }
    m = block_sum(m, red);
    if (m >= target) { lo = mid; kept = m; } else { hi = mid; }
  }

  float inv = 1.f / kept;
  for (int j = threadIdx.x; j < NROWS; j += blockDim.x) {
    float p = prow[j];
    Ab[(size_t)i * NROWS + j] = (p >= lo) ? (bf16)(p * inv) : (bf16)0.f;
  }
}

// ---------------------------------------------------------------------------
// Residual + LayerNorm
// ---------------------------------------------------------------------------
__global__ void final_ln(const float* __restrict__ X, const float* __restrict__ Z,
                         const float* __restrict__ gamma, const float* __restrict__ beta,
                         float* __restrict__ out) {
  __shared__ float red[8];
  int i = blockIdx.x;
  float y[2];
  float s = 0.f;
#pragma unroll
  for (int t = 0; t < 2; ++t) {
    int d = threadIdx.x + t * 256;
    y[t] = X[(size_t)i * DIN + d] + Z[(size_t)i * DIN + d];
    s += y[t];
  }
  s = block_sum(s, red);
  float mu = s * (1.f / DIN);
  float q = 0.f;
#pragma unroll
  for (int t = 0; t < 2; ++t) { float dv = y[t] - mu; q += dv * dv; }
  q = block_sum(q, red);
  float inv = rsqrtf(q * (1.f / DIN) + LNEPS);
#pragma unroll
  for (int t = 0; t < 2; ++t) {
    int d = threadIdx.x + t * 256;
    out[(size_t)i * DIN + d] = (y[t] - mu) * inv * gamma[d] + beta[d];
  }
}

// ---------------------------------------------------------------------------
// Host orchestration
// ---------------------------------------------------------------------------
extern "C" void kernel_launch(void* const* d_in, const int* in_sizes, int n_in,
                              void* d_out, int out_size, void* d_ws, size_t ws_size,
                              hipStream_t stream) {
  (void)in_sizes; (void)n_in; (void)out_size; (void)ws_size;

  const float* X  = (const float*)d_in[0];
  const float* S  = (const float*)d_in[1];
  const float* W1 = (const float*)d_in[2];
  const float* W2 = (const float*)d_in[3];
  const float* W3 = (const float*)d_in[4];
  const float* U1 = (const float*)d_in[5];
  const float* U2 = (const float*)d_in[6];
  const float* U3 = (const float*)d_in[7];
  const float* gamma = (const float*)d_in[8];
  const float* beta  = (const float*)d_in[9];
  float* out = (float*)d_out;

  char* ws = (char*)d_ws;
  size_t off = 0;
  auto alloc = [&](size_t bytes) -> char* {
    char* p = ws + off;
    off += (bytes + 255) & ~(size_t)255;
    return p;
  };

  bf16*  Sb   = (bf16*) alloc((size_t)NROWS * NROWS * 2);
  bf16*  Ab   = (bf16*) alloc((size_t)NROWS * NROWS * 2);
  float* Lg   = (float*)alloc((size_t)NROWS * NROWS * 4);
  float* Tf   = (float*)alloc((size_t)NROWS * DIN * 4);
  float* Z    = (float*)alloc((size_t)NROWS * DIN * 4);
  bf16*  Xf16 = (bf16*) alloc((size_t)NROWS * DIN * 2);
  bf16*  Xb16 = (bf16*) alloc((size_t)NROWS * DIN * 2);
  bf16*  Xa16 = (bf16*) alloc((size_t)NROWS * DIN * 2);
  bf16*  Xf16t= (bf16*) alloc((size_t)DIN * NROWS * 2);
  bf16*  Xb16t= (bf16*) alloc((size_t)DIN * NROWS * 2);
  bf16*  Xa16t= (bf16*) alloc((size_t)DIN * NROWS * 2);
  float* Qf   = (float*)alloc((size_t)NROWS * DATT * 4);
  float* Kf   = (float*)alloc((size_t)NROWS * DATT * 4);
  float* Pf   = (float*)alloc((size_t)NROWS * DATT * 4);
  bf16*  Q16  = (bf16*) alloc((size_t)NROWS * DATT * 2);
  bf16*  K16  = (bf16*) alloc((size_t)NROWS * DATT * 2);
  bf16*  P16  = (bf16*) alloc((size_t)NROWS * DATT * 2);
  bf16*  W1t  = (bf16*) alloc((size_t)DATT * DIN * 2);
  bf16*  W2t  = (bf16*) alloc((size_t)DATT * DATT * 2);
  bf16*  W3t  = (bf16*) alloc((size_t)DATT * DIN * 2);
  bf16*  U1t  = (bf16*) alloc((size_t)NSTEPS * DIN * DIN * 2);
  bf16*  U2t  = (bf16*) alloc((size_t)NSTEPS * DIN * DIN * 2);
  bf16*  U3t  = (bf16*) alloc((size_t)NSTEPS * DIN * DIN * 2);
  float* rowinv = (float*)alloc((size_t)NROWS * 4);
  float* colinv = (float*)alloc((size_t)NROWS * 4);

  // --- Prep ---
  rowprep<<<NROWS, 256, 0, stream>>>(S, Sb, rowinv);
  colprep<<<NROWS / 256, 256, 0, stream>>>(S, colinv);
  init_x<<<NROWS, 256, 0, stream>>>(X, Xf16, Xb16, Xa16, Xf16t, Xb16t, Xa16t, Z);

  tconv_f32_bf16<<<(DIN * DATT + 255) / 256, 256, 0, stream>>>(W1, W1t, DIN, DATT);
  tconv_f32_bf16<<<(DATT * DATT + 255) / 256, 256, 0, stream>>>(W2, W2t, DATT, DATT);
  tconv_f32_bf16<<<(DIN * DATT + 255) / 256, 256, 0, stream>>>(W3, W3t, DIN, DATT);
  for (int s = 0; s < NSTEPS; ++s) {
    size_t o = (size_t)s * DIN * DIN;
    tconv_f32_bf16<<<(DIN * DIN + 255) / 256, 256, 0, stream>>>(U1 + o, U1t + o, DIN, DIN);
    tconv_f32_bf16<<<(DIN * DIN + 255) / 256, 256, 0, stream>>>(U2 + o, U2t + o, DIN, DIN);
    tconv_f32_bf16<<<(DIN * DIN + 255) / 256, 256, 0, stream>>>(U3 + o, U3t + o, DIN, DIN);
  }

  dim3 blkS(128);
  dim3 blkL(256);
  dim3 gAtt(DATT / 64, NROWS / 128);    // (1, 32)  direct-global GEMM
  dim3 gLg(NROWS / 128, NROWS / 128);   // (32, 32) LDS-async GEMM
  dim3 gT(DIN / 128, NROWS / 128);      // (4, 32)  LDS-async GEMM

  for (int s = 0; s < NSTEPS; ++s) {
    // --- attention adjacency from current Xa ---
    gemm_bf16<<<gAtt, blkS, 0, stream>>>(Xa16, DIN, W1t, DIN, Qf, DATT, DIN, 0);
    gemm_bf16<<<gAtt, blkS, 0, stream>>>(Xa16, DIN, W3t, DIN, Kf, DATT, DIN, 0);
    conv_f32_bf16<<<(NROWS * DATT + 255) / 256, 256, 0, stream>>>(Qf, Q16, NROWS * DATT);
    conv_f32_bf16<<<(NROWS * DATT + 255) / 256, 256, 0, stream>>>(Kf, K16, NROWS * DATT);
    gemm_bf16<<<gAtt, blkS, 0, stream>>>(Q16, DATT, W2t, DATT, Pf, DATT, DATT, 0);
    conv_f32_bf16<<<(NROWS * DATT + 255) / 256, 256, 0, stream>>>(Pf, P16, NROWS * DATT);
    // logits = P @ K^T  (tau == 1.0);  Bt for K^T is K row-major
    gemm_bf16_lds<<<gLg, blkL, 0, stream>>>(P16, DATT, K16, DATT, Lg, NROWS, DATT, 0);
    nucleus_kernel<<<NROWS, 256, 0, stream>>>(Lg, Ab);

    // --- diffusion updates: Xf = Rf@Xf, Xb = Rb@Xb, Xa = A@Xa ---
    gemm_bf16_lds<<<gT, blkL, 0, stream>>>(Sb, NROWS, Xf16t, NROWS, Tf, DIN, NROWS, 0);
    scale_conv<<<NROWS, 256, 0, stream>>>(Tf, rowinv, Xf16, Xf16t);
    gemm_bf16_lds<<<gT, blkL, 0, stream>>>(Sb, NROWS, Xb16t, NROWS, Tf, DIN, NROWS, 0);
    scale_conv<<<NROWS, 256, 0, stream>>>(Tf, colinv, Xb16, Xb16t);
    gemm_bf16_lds<<<gT, blkL, 0, stream>>>(Ab, NROWS, Xa16t, NROWS, Tf, DIN, NROWS, 0);
    scale_conv<<<NROWS, 256, 0, stream>>>(Tf, (const float*)nullptr, Xa16, Xa16t);

    // --- Z += Xf@U1[s] + Xb@U2[s] + Xa@U3[s] ---
    size_t o = (size_t)s * DIN * DIN;
    gemm_bf16_lds<<<gT, blkL, 0, stream>>>(Xf16, DIN, U1t + o, DIN, Z, DIN, DIN, 1);
    gemm_bf16_lds<<<gT, blkL, 0, stream>>>(Xb16, DIN, U2t + o, DIN, Z, DIN, DIN, 1);
    gemm_bf16_lds<<<gT, blkL, 0, stream>>>(Xa16, DIN, U3t + o, DIN, Z, DIN, DIN, 1);
  }

  // --- Y = X + Z ; LayerNorm ---
  final_ln<<<NROWS, 256, 0, stream>>>(X, Z, gamma, beta, out);
}
